// AdaptiveEncoder_45921790329007
// MI455X (gfx1250) — matrix operand order repaired and checked by reading, not compiled, not run on hardware
//
#include <hip/hip_runtime.h>
#include <hip/hip_fp16.h>

typedef __attribute__((ext_vector_type(16))) _Float16 v16h;
typedef __attribute__((ext_vector_type(8)))  _Float16 v8h;
typedef __attribute__((ext_vector_type(8)))  float    v8f;

#define EPSBN 1e-5f
#define GRID(n) ((unsigned)(((n) + 255) / 256))

__device__ __forceinline__ float sigmoidf_(float x) { return 1.f / (1.f + expf(-x)); }

// WMMA A-fragment K index for element i, half-wave hi (ISA 7.12.2)
__device__ __forceinline__ int kA_idx(int i, int hi) { return (i & 7) + ((i >> 3) << 4) + (hi << 3); }

// ---------------- utility kernels ----------------
__global__ void k_zero(float* p, int n) {
  int i = blockIdx.x * blockDim.x + threadIdx.x;
  if (i < n) p[i] = 0.f;
}

__global__ void k_f32_to_f16(const float* __restrict__ x, _Float16* __restrict__ y, int n) {
  int i = blockIdx.x * blockDim.x + threadIdx.x;
  if (i < n) y[i] = (_Float16)x[i];
}

// pack conv weights (OIDHW f32) into WMMA A-fragment order: [mt][tap][cb][lane][16]
__global__ void k_prepack_convw(const float* __restrict__ W, _Float16* __restrict__ out,
                                int Ci, int Co) {
  int idx = blockIdx.x * blockDim.x + threadIdx.x;
  int total = Co * Ci * 27;
  if (idx >= total) return;
  int i = idx & 15;
  int lane = (idx >> 4) & 31;
  int f = idx >> 9;
  int kchunks = Ci >> 5;
  int cb = f % kchunks;
  int rest = f / kchunks;
  int tap = rest % 27;
  int mt = rest / 27;
  int hi = lane >> 4, lr = lane & 15;
  int co = mt * 16 + lr;
  int ci = (cb << 5) + i + (hi << 4);
  out[idx] = (_Float16)W[((size_t)(co * Ci) + ci) * 27 + tap];
}

// pack concatenated LSTM gate weights (1280 x 512) into WMMA B-fragment order: [jt*16+cb][lane][16]
__global__ void k_prepack_gatesw(const float* __restrict__ wih, const float* __restrict__ whh,
                                 const float* __restrict__ wxg, const float* __restrict__ whg,
                                 _Float16* __restrict__ out) {
  int idx = blockIdx.x * blockDim.x + threadIdx.x;
  if (idx >= 80 * 16 * 512) return;
  int i = idx & 15;
  int lane = (idx >> 4) & 31;
  int f = idx >> 9;           // jt*16 + cb
  int cb = f & 15;
  int jt = f >> 4;
  int hi = lane >> 4, lr = lane & 15;
  int j = jt * 16 + lr;
  int k = (cb << 5) + i + (hi << 4);   // B-fragment K pattern
  float w;
  if (j < 1024) w = (k < 256) ? wih[(size_t)j * 256 + k] : whh[(size_t)j * 256 + (k - 256)];
  else { int jj = j - 1024;
         w = (k < 256) ? wxg[(size_t)jj * 256 + k] : whg[(size_t)jj * 256 + (k - 256)]; }
  out[idx] = (_Float16)w;
}

// ---------------- conv1 direct (Cin=4), writes channels-last f16 ----------------
__global__ void k_conv3d_direct(const float* __restrict__ X, const float* __restrict__ W,
                                const float* __restrict__ bias, _Float16* __restrict__ Y,
                                int B, int Ci, int Co, int Din, int Dout) {
  int idx = blockIdx.x * blockDim.x + threadIdx.x;
  int D3 = Dout * Dout * Dout;
  int total = B * D3 * Co;
  if (idx >= total) return;
  int co = idx % Co; int r = idx / Co; int sp = r % D3; int b = r / D3;
  int od = sp / (Dout * Dout); int rr = sp % (Dout * Dout); int oh = rr / Dout; int ow = rr % Dout;
  int Din2 = Din * Din;
  float acc = bias[co];
  for (int ci = 0; ci < Ci; ++ci) {
    const float* xb = X + ((size_t)(b * Ci + ci)) * Din2 * Din;
    const float* wb = W + ((size_t)(co * Ci + ci)) * 27;
    for (int kd = 0; kd < 3; ++kd) { int id = od * 2 - 1 + kd; if (id < 0 || id >= Din) continue;
      for (int kh = 0; kh < 3; ++kh) { int ih = oh * 2 - 1 + kh; if (ih < 0 || ih >= Din) continue;
        for (int kw = 0; kw < 3; ++kw) { int iw = ow * 2 - 1 + kw; if (iw < 0 || iw >= Din) continue;
          acc += xb[id * Din2 + ih * Din + iw] * wb[(kd * 3 + kh) * 3 + kw];
        } } }
  }
  acc = acc > 0.f ? acc : 0.f;
  Y[idx] = (_Float16)acc;   // (b, sp, co) channels-last
}

// ---------------- BN for channels-last tensors: coalesced 2-stage reduction ----------------
template <typename T>
__global__ void k_bnstats_cl_partial(const T* __restrict__ X, int N, int C,
                                     float* __restrict__ ps, float* __restrict__ pq) {
  int tid = threadIdx.x;           // 256 threads, C power of two <= 256
  int c = tid & (C - 1);
  int g = tid / C;
  int G = 256 / C;
  float a = 0.f, q = 0.f;
  for (int p = blockIdx.x * G + g; p < N; p += gridDim.x * G) {
    float v = (float)X[(size_t)p * C + c];
    a += v; q += v * v;
  }
  __shared__ float s1[256], s2[256];
  s1[tid] = a; s2[tid] = q;
  __syncthreads();
  if (tid < C) {
    for (int k = 1; k < G; ++k) { s1[tid] += s1[tid + k * C]; s2[tid] += s2[tid + k * C]; }
    ps[blockIdx.x * C + tid] = s1[tid];
    pq[blockIdx.x * C + tid] = s2[tid];
  }
}

__global__ void k_bnstats_cl_final(const float* __restrict__ ps, const float* __restrict__ pq,
                                   int NB, int C, int N, float* __restrict__ mean, float* __restrict__ istd) {
  int c = blockIdx.x * blockDim.x + threadIdx.x;
  if (c >= C) return;
  float a = 0.f, q = 0.f;
  for (int k = 0; k < NB; ++k) { a += ps[k * C + c]; q += pq[k * C + c]; }
  float m = a / N;
  float v = q / N - m * m;
  mean[c] = m; istd[c] = rsqrtf(v + EPSBN);
}

template <typename T>
__global__ void k_bnapply_cl(T* __restrict__ X, const float* __restrict__ g, const float* __restrict__ be,
                             const float* __restrict__ mean, const float* __restrict__ istd,
                             int C, int total) {
  int i = blockIdx.x * blockDim.x + threadIdx.x;
  if (i >= total) return;
  int c = i & (C - 1);
  float v = (float)X[i];
  X[i] = (T)(g[c] * (v - mean[c]) * istd[c] + be[c]);
}

// BN-apply for V: normalize in place (channels-last) AND scatter Vh into WMMA A-fragment order
// V is (64 b, 64 n, 256 c); fragments: f = (b*4 + n/16)*8 + cb, lane = hi*16 + (n&15)
__global__ void k_bnapply_V(float* __restrict__ V, const float* __restrict__ g, const float* __restrict__ be,
                            const float* __restrict__ mean, const float* __restrict__ istd,
                            _Float16* __restrict__ Vf) {
  int i = blockIdx.x * blockDim.x + threadIdx.x;
  if (i >= 64 * 64 * 256) return;
  int c = i & 255;
  float v = V[i];
  v = g[c] * (v - mean[c]) * istd[c] + be[c];
  V[i] = v;
  int n = (i >> 8) & 63;
  int b = i >> 14;
  int cb = c >> 5;
  int kk = c & 31;                       // contiguous K: kk = 16*hi + ii
  int hi = kk >> 4;
  int ii = kk & 15;
  int lane = (hi << 4) + (n & 15);
  int f = (b * 4 + (n >> 4)) * 8 + cb;
  Vf[((size_t)f << 9) + (lane << 4) + ii] = (_Float16)v;
}

// ---------------- BN channel-major (small text tensors) ----------------
template <typename T>
__global__ void k_bnstats(const T* __restrict__ X, int B, int C, int S,
                          float* __restrict__ mean, float* __restrict__ istd) {
  int c = blockIdx.x;
  __shared__ float s1[256], s2[256];
  float a = 0.f, q = 0.f;
  int n = B * S;
  for (int i = threadIdx.x; i < n; i += blockDim.x) {
    int bb = i / S, s = i - bb * S;
    float v = (float)X[((size_t)(bb * C + c)) * S + s];
    a += v; q += v * v;
  }
  s1[threadIdx.x] = a; s2[threadIdx.x] = q;
  __syncthreads();
  for (int off = 128; off > 0; off >>= 1) {
    if ((int)threadIdx.x < off) { s1[threadIdx.x] += s1[threadIdx.x + off]; s2[threadIdx.x] += s2[threadIdx.x + off]; }
    __syncthreads();
  }
  if (threadIdx.x == 0) {
    float m = s1[0] / n;
    float v = s2[0] / n - m * m;
    mean[c] = m; istd[c] = rsqrtf(v + EPSBN);
  }
}

template <typename T>
__global__ void k_bnapply(T* __restrict__ X, const float* __restrict__ g, const float* __restrict__ be,
                          const float* __restrict__ mean, const float* __restrict__ istd,
                          int C, int S, int total) {
  int i = blockIdx.x * blockDim.x + threadIdx.x;
  if (i >= total) return;
  int c = (i / S) % C;
  float v = (float)X[i];
  X[i] = (T)(g[c] * (v - mean[c]) * istd[c] + be[c]);
}

// ---------------- implicit-GEMM conv3d (s2 p1 k3), channels-last, 2x2 tiles/wave ----------------
// Low register pressure: one 32-bit base offset per position + 27-bit validity mask,
// per-tap delta is wave-uniform (SALU).
__global__ void k_conv3d_wmma(const _Float16* __restrict__ X, const _Float16* __restrict__ Wp,
                              const float* __restrict__ bias,
                              _Float16* __restrict__ Yh, float* __restrict__ Yf,
                              int B, int Ci, int Co, int Din, int Dout) {
  const int lane = threadIdx.x & 31;
  const int wid = blockIdx.x * (blockDim.x >> 5) + (threadIdx.x >> 5);
  const int D3 = Dout * Dout * Dout;
  const int npos = B * D3;
  const int ntM2 = Co >> 5;
  const int ntN2 = npos >> 5;
  if (wid >= ntM2 * ntN2) return;   // whole-wave uniform -> EXEC all-ones at WMMA
  const int mt2 = wid % ntM2;
  const int nt2 = wid / ntM2;
  const int co0 = mt2 << 5;
  const int hi = lane >> 4;
  const int lr = lane & 15;
  const int kchunks = Ci >> 5;
  int pos[2], base_off[2];
  unsigned okmask[2];
#pragma unroll
  for (int tp = 0; tp < 2; ++tp) {
    int p = (nt2 << 5) + (tp << 4) + lr;
    pos[tp] = p;
    int b = p / D3;
    int sp = p - b * D3;
    int od = sp / (Dout * Dout); sp -= od * Dout * Dout;
    int oh = sp / Dout;
    int ow = sp - oh * Dout;
    int id0 = od * 2 - 1, ih0 = oh * 2 - 1, iw0 = ow * 2 - 1;
    base_off[tp] = (((b * Din + id0) * Din + ih0) * Din + iw0) * Ci;
    unsigned m = 0;
#pragma unroll
    for (int kd = 0; kd < 3; ++kd)
#pragma unroll
      for (int kh = 0; kh < 3; ++kh)
#pragma unroll
        for (int kw = 0; kw < 3; ++kw) {
          bool vv = ((unsigned)(id0 + kd) < (unsigned)Din) &&
                    ((unsigned)(ih0 + kh) < (unsigned)Din) &&
                    ((unsigned)(iw0 + kw) < (unsigned)Din);
          m |= (vv ? 1u : 0u) << ((kd * 3 + kh) * 3 + kw);
        }
    okmask[tp] = m;
  }
  v8f acc00 = {}, acc01 = {}, acc10 = {}, acc11 = {};
  v16h bz = {};
  const _Float16* wlane = Wp + (lane << 4);
#pragma unroll 1
  for (int tap = 0; tap < 27; ++tap) {
    int kd = tap / 9;
    int rem = tap - kd * 9;
    int kh = rem / 3;
    int kw = rem - kh * 3;
    int delta = ((kd * Din + kh) * Din + kw) * Ci;      // wave-uniform
    bool ok0 = (okmask[0] >> tap) & 1;
    bool ok1 = (okmask[1] >> tap) & 1;
    int f0 = ((mt2 * 2) * 27 + tap) * kchunks;
    int f1 = f0 + 27 * kchunks;
#pragma unroll 1
    for (int cb = 0; cb < kchunks; ++cb) {
      v16h a0 = *(const v16h*)(wlane + (((size_t)(f0 + cb)) << 9));
      v16h a1 = *(const v16h*)(wlane + (((size_t)(f1 + cb)) << 9));
      int koff = delta + (cb << 5) + (hi << 4);
      v16h bA = ok0 ? *(const v16h*)(X + base_off[0] + koff) : bz;
      v16h bB = ok1 ? *(const v16h*)(X + base_off[1] + koff) : bz;
      acc00 = __builtin_amdgcn_wmma_f32_16x16x32_f16(false, a0, false, bA, (short)0, acc00, false, false);
      acc01 = __builtin_amdgcn_wmma_f32_16x16x32_f16(false, a0, false, bB, (short)0, acc01, false, false);
      acc10 = __builtin_amdgcn_wmma_f32_16x16x32_f16(false, a1, false, bA, (short)0, acc10, false, false);
      acc11 = __builtin_amdgcn_wmma_f32_16x16x32_f16(false, a1, false, bB, (short)0, acc11, false, false);
    }
  }
  v8f accs[2][2] = {{acc00, acc01}, {acc10, acc11}};
#pragma unroll
  for (int tm = 0; tm < 2; ++tm) {
    int cobase = co0 + (tm << 4) + (hi << 3);
#pragma unroll
    for (int tp = 0; tp < 2; ++tp) {
      size_t base = (size_t)pos[tp] * Co + cobase;
      if (Yh) {
        v8h y;
#pragma unroll
        for (int r = 0; r < 8; ++r) { float v = accs[tm][tp][r] + bias[cobase + r]; y[r] = (_Float16)(v > 0.f ? v : 0.f); }
        *(v8h*)(Yh + base) = y;
      }
      if (Yf) {
        v8f y;
#pragma unroll
        for (int r = 0; r < 8; ++r) { float v = accs[tm][tp][r] + bias[cobase + r]; y[r] = v > 0.f ? v : 0.f; }
        *(v8f*)(Yf + base) = y;
      }
    }
  }
}

// ---------------- vproj = einsum('bcn,oc->bno', V, wv) + bv ----------------
// Vf: A-fragment-packed V (f16); wv row-major (512,256) f16
__global__ void k_vproj_wmma(const _Float16* __restrict__ Vf, const _Float16* __restrict__ Wvh,
                             const float* __restrict__ bv, float* __restrict__ vproj, int B) {
  const int lane = threadIdx.x & 31;
  const int wid = blockIdx.x * (blockDim.x >> 5) + (threadIdx.x >> 5);
  if (wid >= B * 128) return;
  const int b = wid >> 7;
  const int t = wid & 127;
  const int n0 = (t & 3) << 4;
  const int o0 = (t >> 2) << 4;
  const int hi = lane >> 4;
  const int lr = lane & 15;
  v8f acc = {};
  for (int cb = 0; cb < 8; ++cb) {
    int f = (b * 4 + (n0 >> 4)) * 8 + cb;
    v16h a = *(const v16h*)(Vf + ((size_t)f << 9) + (lane << 4));
    v16h bf = *(const v16h*)(Wvh + (size_t)(o0 + lr) * 256 + (cb << 5) + (hi << 4));
    acc = __builtin_amdgcn_wmma_f32_16x16x32_f16(false, a, false, bf, (short)0, acc, false, false);
  }
#pragma unroll
  for (int r = 0; r < 8; ++r) {
    int n = n0 + r + (hi << 3);
    int o = o0 + lr;
    vproj[((size_t)b * 64 + n) * 512 + o] = acc[r] + bv[o];
  }
}

// ---------------- LSTM gate GEMM: [x_t ; h] (64x512) x Wcat^T -> (64x1280) ----------------
// Af: A-fragment-packed [x_t;h]; Wgp: B-fragment-packed weights
__global__ void k_gates_wmma(const _Float16* __restrict__ Af, const _Float16* __restrict__ Wgp,
                             const float* __restrict__ bih, const float* __restrict__ bhh,
                             const float* __restrict__ bxg, const float* __restrict__ bhg,
                             float* __restrict__ G) {
  const int lane = threadIdx.x & 31;
  const int wid = blockIdx.x * (blockDim.x >> 5) + (threadIdx.x >> 5);
  if (wid >= 320) return;
  const int bt = wid & 3;
  const int jt = wid >> 2;           // 80 tiles of 16 outputs
  const int hi = lane >> 4;
  const int lr = lane & 15;
  const int j = jt * 16 + lr;
  v8f acc = {};
  for (int cb = 0; cb < 16; ++cb) {
    v16h a = *(const v16h*)(Af + (((size_t)(bt * 16 + cb)) << 9) + (lane << 4));
    v16h bf = *(const v16h*)(Wgp + (((size_t)(jt * 16 + cb)) << 9) + (lane << 4));
    acc = __builtin_amdgcn_wmma_f32_16x16x32_f16(false, a, false, bf, (short)0, acc, false, false);
  }
  float bias = (j < 1024) ? (bih[j] + bhh[j]) : (bxg[j - 1024] + bhg[j - 1024]);
#pragma unroll
  for (int r = 0; r < 8; ++r) {
    int b = (bt << 4) + r + (hi << 3);
    G[(size_t)b * 1280 + j] = acc[r] + bias;
  }
}

// ---------------- text path ----------------
__global__ void k_embed(const int* __restrict__ ids, const float* __restrict__ emb,
                        float* __restrict__ out, int B, int T, int E) {
  int i = blockIdx.x * blockDim.x + threadIdx.x;
  int total = B * E * T;
  if (i >= total) return;
  int t = i % T; int r = i / T; int e = r % E; int b = r / E;
  out[i] = emb[(size_t)ids[b * T + t] * E + e];
}

__global__ void k_conv1d(const float* __restrict__ X, const float* __restrict__ W,
                         const float* __restrict__ bias, float* __restrict__ Y,
                         int B, int Ci, int Co, int T) {
  int i = blockIdx.x * blockDim.x + threadIdx.x;
  int total = B * Co * T;
  if (i >= total) return;
  int t = i % T; int r = i / T; int co = r % Co; int b = r / Co;
  float acc = bias ? bias[co] : 0.f;
  for (int ci = 0; ci < Ci; ++ci) {
    const float* xb = X + (size_t)(b * Ci + ci) * T;
    const float* wb = W + (size_t)(co * Ci + ci) * 3;
    for (int kk = 0; kk < 3; ++kk) { int tt = t - 1 + kk; if (tt >= 0 && tt < T) acc += xb[tt] * wb[kk]; }
  }
  Y[i] = acc > 0.f ? acc : 0.f;
}

// ---------------- per-step scan kernels ----------------
// Pack [x_t ; h] directly into WMMA A-fragment order: [bt*16+cb][lane][16]
__global__ void k_pack_frag(const float* __restrict__ tf, const float* __restrict__ h,
                            _Float16* __restrict__ Af, int step, int B) {
  int idx = blockIdx.x * blockDim.x + threadIdx.x;
  if (idx >= 64 * 512) return;
  int i = idx & 15;
  int lane = (idx >> 4) & 31;
  int f = idx >> 9;            // bt*16 + cb
  int cb = f & 15;
  int bt = f >> 4;
  int hi = lane >> 4, lr = lane & 15;
  int row = bt * 16 + lr;      // batch index
  int k = (cb << 5) + i + (hi << 4);
  float v = (k < 256) ? tf[((size_t)(row * 256 + k)) * 32 + step] : h[row * 256 + (k - 256)];
  Af[idx] = (_Float16)v;
}

__global__ void k_state(const float* __restrict__ G, float* __restrict__ c,
                        float* __restrict__ hnew, float* __restrict__ sent, int B) {
  int i = blockIdx.x * blockDim.x + threadIdx.x;
  if (i >= B * 256) return;
  int j = i % 256, b = i / 256;
  const float* g = G + (size_t)b * 1280;
  float ig = sigmoidf_(g[j]);
  float fg = sigmoidf_(g[256 + j]);
  float gg = tanhf(g[512 + j]);
  float og = sigmoidf_(g[768 + j]);
  float xg = sigmoidf_(g[1024 + j]);
  float cn = fg * c[i] + ig * gg;
  float tc = tanhf(cn);
  c[i] = cn;
  hnew[i] = og * tc;
  sent[i] = xg * tc;
}

__global__ void k_hp(const float* __restrict__ hnew, const float* __restrict__ wh,
                     const float* __restrict__ bh, float* __restrict__ hp, int B) {
  int i = blockIdx.x * blockDim.x + threadIdx.x;
  if (i >= B * 512) return;
  int o = i % 512, b = i / 512;
  const float* hb = hnew + b * 256;
  const float* w = wh + (size_t)o * 256;
  float acc = bh[o];
  for (int cc = 0; cc < 256; ++cc) acc += hb[cc] * w[cc];
  hp[i] = acc;
}

__global__ void k_sz(const float* __restrict__ sent, const float* __restrict__ wsw,
                     const float* __restrict__ bs, const float* __restrict__ hp,
                     const float* __restrict__ wo, const float* __restrict__ bo,
                     float* __restrict__ sz, int B) {
  int b = blockIdx.x;
  int tid = threadIdx.x;  // 256
  __shared__ float red[256];
  const float* sb = sent + b * 256;
  float partial = 0.f;
  for (int o = tid; o < 512; o += 256) {
    const float* w = wsw + (size_t)o * 256;
    float acc = bs[o] + hp[b * 512 + o];
    for (int cc = 0; cc < 256; ++cc) acc += sb[cc] * w[cc];
    partial += tanhf(acc) * wo[o];
  }
  red[tid] = partial; __syncthreads();
  for (int off = 128; off > 0; off >>= 1) { if (tid < off) red[tid] += red[tid + off]; __syncthreads(); }
  if (tid == 0) sz[b] = red[0] + bo[0];
}

__global__ void k_zatt(const float* __restrict__ vproj, const float* __restrict__ hp,
                       const float* __restrict__ wo, const float* __restrict__ bo,
                       float* __restrict__ z, int B) {
  int bn = blockIdx.x;        // b*64 + n
  int b = bn >> 6;
  int tid = threadIdx.x;      // 64
  __shared__ float red[64];
  const float* vp = vproj + (size_t)bn * 512;
  const float* hb = hp + b * 512;
  float partial = 0.f;
  for (int o = tid; o < 512; o += 64) partial += tanhf(vp[o] + hb[o]) * wo[o];
  red[tid] = partial; __syncthreads();
  for (int off = 32; off > 0; off >>= 1) { if (tid < off) red[tid] += red[tid + off]; __syncthreads(); }
  if (tid == 0) z[bn] = red[0] + bo[0];
}

// V is channels-last (B, 64pos, 256c)
__global__ void k_attend(const float* __restrict__ z, const float* __restrict__ sz,
                         const float* __restrict__ V, const float* __restrict__ sent,
                         const float* __restrict__ hnew, float* __restrict__ h,
                         float* __restrict__ ctx_sum, int B) {
  int b = blockIdx.x;
  int tid = threadIdx.x;  // 256
  __shared__ float sm[66];
  if (tid < 64) sm[tid] = z[b * 64 + tid];
  if (tid == 64) sm[64] = sz[b];
  __syncthreads();
  if (tid == 0) {
    float m = sm[0];
    for (int k = 1; k < 65; ++k) m = sm[k] > m ? sm[k] : m;
    float s = 0.f;
    for (int k = 0; k < 65; ++k) { float e = expf(sm[k] - m); sm[k] = e; s += e; }
    sm[65] = s;
  }
  __syncthreads();
  float inv = 1.f / sm[65];
  float beta = sm[64] * inv;
  int cix = b * 256 + tid;
  float att = 0.f;
  for (int n = 0; n < 64; ++n) att += V[((size_t)(b * 64 + n)) * 256 + tid] * sm[n];
  att *= inv;
  float ctx = (1.f - beta) * att;
  ctx_sum[cix] += ctx;
  h[cix] = beta * sent[cix] + ctx + hnew[cix];
}

// ---------------- output heads ----------------
__global__ void k_shapeout(const float* __restrict__ ctx_sum, const float* __restrict__ sow,
                           const float* __restrict__ sob, float* __restrict__ out, int B) {
  int i = blockIdx.x * blockDim.x + threadIdx.x;
  if (i >= B * 128) return;
  int o = i % 128, b = i / 128;
  const float* a = ctx_sum + b * 256;
  const float* w = sow + (size_t)o * 256;
  float acc = 0.f;
  for (int cc = 0; cc < 256; ++cc) acc += a[cc] * w[cc];
  out[i] = acc * (1.f / 32.f) + sob[o];
}

__global__ void k_texthidden(const float* __restrict__ h, const float* __restrict__ tow1,
                             const float* __restrict__ tob1, float* __restrict__ th, int B) {
  int i = blockIdx.x * blockDim.x + threadIdx.x;
  if (i >= B * 256) return;
  int j = i % 256, b = i / 256;
  const float* hb = h + b * 256;
  const float* w = tow1 + (size_t)j * 256;
  float acc = tob1[j];
  for (int cc = 0; cc < 256; ++cc) acc += hb[cc] * w[cc];
  th[i] = acc > 0.f ? acc : 0.f;
}

__global__ void k_textout(const float* __restrict__ th, const float* __restrict__ tow2,
                          const float* __restrict__ tob2, float* __restrict__ out, int B) {
  int i = blockIdx.x * blockDim.x + threadIdx.x;
  if (i >= B * 128) return;
  int o = i % 128, b = i / 128;
  const float* tb = th + b * 256;
  const float* w = tow2 + (size_t)o * 256;
  float acc = tob2[o];
  for (int cc = 0; cc < 256; ++cc) acc += tb[cc] * w[cc];
  out[i] = acc;
}

// ---------------- host launch ----------------
extern "C" void kernel_launch(void* const* d_in, const int* in_sizes, int n_in,
                              void* d_out, int out_size, void* d_ws, size_t ws_size,
                              hipStream_t stream) {
  (void)in_sizes; (void)n_in; (void)out_size; (void)ws_size;
  const float* shape_in = (const float*)d_in[0];
  const int*   text_in  = (const int*)d_in[1];
  const float* c1w = (const float*)d_in[2];  const float* c1b = (const float*)d_in[3];
  const float* g1  = (const float*)d_in[4];  const float* be1 = (const float*)d_in[5];
  const float* c2w = (const float*)d_in[6];  const float* c2b = (const float*)d_in[7];
  const float* g2  = (const float*)d_in[8];  const float* be2 = (const float*)d_in[9];
  const float* c3w = (const float*)d_in[10]; const float* c3b = (const float*)d_in[11];
  const float* g3  = (const float*)d_in[12]; const float* be3 = (const float*)d_in[13];
  const float* sow = (const float*)d_in[14]; const float* sob = (const float*)d_in[15];
  const float* emb = (const float*)d_in[16];
  const float* t1w = (const float*)d_in[17]; const float* t1b = (const float*)d_in[18];
  const float* t2w = (const float*)d_in[19];
  const float* g128 = (const float*)d_in[20]; const float* b128 = (const float*)d_in[21];
  const float* t3w = (const float*)d_in[22]; const float* t3b = (const float*)d_in[23];
  const float* t4w = (const float*)d_in[24];
  const float* g256 = (const float*)d_in[25]; const float* b256 = (const float*)d_in[26];
  const float* tow1 = (const float*)d_in[27]; const float* tob1 = (const float*)d_in[28];
  const float* tow2 = (const float*)d_in[29]; const float* tob2 = (const float*)d_in[30];
  const float* wih = (const float*)d_in[31]; const float* whh = (const float*)d_in[32];
  const float* bih = (const float*)d_in[33]; const float* bhh = (const float*)d_in[34];
  const float* wxg = (const float*)d_in[35]; const float* bxg = (const float*)d_in[36];
  const float* whg = (const float*)d_in[37]; const float* bhg = (const float*)d_in[38];
  const float* wv  = (const float*)d_in[39]; const float* bv  = (const float*)d_in[40];
  const float* wh  = (const float*)d_in[41]; const float* bh  = (const float*)d_in[42];
  const float* wsw = (const float*)d_in[43]; const float* bs  = (const float*)d_in[44];
  const float* wo  = (const float*)d_in[45]; const float* bo  = (const float*)d_in[46];

  char* ws = (char*)d_ws;
  size_t off = 0;
  auto carve = [&](size_t bytes) -> char* {
    char* p = ws + off;
    off += (bytes + 255) & ~(size_t)255;
    return p;
  };
  _Float16* x1h   = (_Float16*)carve((size_t)16777216 * 2);  // (64, 16^3, 64) channels-last
  _Float16* x2h   = (_Float16*)carve((size_t)4194304 * 2);   // (64, 8^3, 128) channels-last
  float*    V     = (float*)carve((size_t)1048576 * 4);      // (64, 64, 256) channels-last
  _Float16* Vf    = (_Float16*)carve((size_t)1048576 * 2);   // V in A-fragment order
  float*    vproj = (float*)carve((size_t)2097152 * 4);      // (64, 64, 512)
  _Float16* c2wp  = (_Float16*)carve((size_t)221184 * 2);    // packed WMMA A-fragments
  _Float16* c3wp  = (_Float16*)carve((size_t)884736 * 2);
  _Float16* wv_h  = (_Float16*)carve((size_t)131072 * 2);
  _Float16* Wgp   = (_Float16*)carve((size_t)655360 * 2);    // packed gate B-fragments
  float* temb = (float*)carve((size_t)262144 * 4);
  float* ta   = (float*)carve((size_t)524288 * 4);
  float* tb   = (float*)carve((size_t)524288 * 4);
  float* meanb = (float*)carve(256 * 4);
  float* istdb = (float*)carve(256 * 4);
  float* psum  = (float*)carve((size_t)128 * 256 * 4);
  float* psq   = (float*)carve((size_t)128 * 256 * 4);
  _Float16* Af  = (_Float16*)carve((size_t)32768 * 2);       // [x_t;h] in A-fragment order
  float* G    = (float*)carve((size_t)81920 * 4);
  float* hbuf = (float*)carve((size_t)16384 * 4);
  float* cbuf = (float*)carve((size_t)16384 * 4);
  float* hnew = (float*)carve((size_t)16384 * 4);
  float* sent = (float*)carve((size_t)16384 * 4);
  float* hp   = (float*)carve((size_t)32768 * 4);
  float* szb  = (float*)carve(64 * 4);
  float* zb   = (float*)carve((size_t)4096 * 4);
  float* ctxs = (float*)carve((size_t)16384 * 4);
  float* th   = (float*)carve((size_t)16384 * 4);

  float* out_f = (float*)d_out;
  const int B = 64;
  const int NB = 128;

  k_zero<<<GRID(16384), 256, 0, stream>>>(hbuf, 16384);
  k_zero<<<GRID(16384), 256, 0, stream>>>(cbuf, 16384);
  k_zero<<<GRID(16384), 256, 0, stream>>>(ctxs, 16384);

  // --- 3D conv tower (channels-last activations) ---
  k_conv3d_direct<<<GRID(16777216), 256, 0, stream>>>(shape_in, c1w, c1b, x1h, B, 4, 64, 32, 16);
  k_bnstats_cl_partial<_Float16><<<NB, 256, 0, stream>>>(x1h, 262144, 64, psum, psq);
  k_bnstats_cl_final<<<1, 256, 0, stream>>>(psum, psq, NB, 64, 262144, meanb, istdb);
  k_bnapply_cl<_Float16><<<GRID(16777216), 256, 0, stream>>>(x1h, g1, be1, meanb, istdb, 64, 16777216);

  k_prepack_convw<<<GRID(221184), 256, 0, stream>>>(c2w, c2wp, 64, 128);
  k_conv3d_wmma<<<512, 256, 0, stream>>>(x1h, c2wp, c2b, x2h, (float*)nullptr, B, 64, 128, 16, 8);
  k_bnstats_cl_partial<_Float16><<<NB, 256, 0, stream>>>(x2h, 32768, 128, psum, psq);
  k_bnstats_cl_final<<<1, 256, 0, stream>>>(psum, psq, NB, 128, 32768, meanb, istdb);
  k_bnapply_cl<_Float16><<<GRID(4194304), 256, 0, stream>>>(x2h, g2, be2, meanb, istdb, 128, 4194304);

  k_prepack_convw<<<GRID(884736), 256, 0, stream>>>(c3w, c3wp, 128, 256);
  k_conv3d_wmma<<<128, 256, 0, stream>>>(x2h, c3wp, c3b, (_Float16*)nullptr, V, B, 128, 256, 8, 4);
  k_bnstats_cl_partial<float><<<NB, 256, 0, stream>>>(V, 4096, 256, psum, psq);
  k_bnstats_cl_final<<<1, 256, 0, stream>>>(psum, psq, NB, 256, 4096, meanb, istdb);
  k_bnapply_V<<<GRID(1048576), 256, 0, stream>>>(V, g3, be3, meanb, istdb, Vf);

  // vproj
  k_f32_to_f16<<<GRID(131072), 256, 0, stream>>>(wv, wv_h, 131072);
  k_vproj_wmma<<<1024, 256, 0, stream>>>(Vf, wv_h, bv, vproj, B);

  // --- text tower (channel-major, tiny) ---
  k_embed<<<GRID(262144), 256, 0, stream>>>(text_in, emb, temb, B, 32, 128);
  k_conv1d<<<GRID(262144), 256, 0, stream>>>(temb, t1w, t1b, ta, B, 128, 128, 32);
  k_conv1d<<<GRID(262144), 256, 0, stream>>>(ta, t2w, (const float*)nullptr, tb, B, 128, 128, 32);
  k_bnstats<float><<<128, 256, 0, stream>>>(tb, B, 128, 32, meanb, istdb);
  k_bnapply<float><<<GRID(262144), 256, 0, stream>>>(tb, g128, b128, meanb, istdb, 128, 32, 262144);
  k_conv1d<<<GRID(524288), 256, 0, stream>>>(tb, t3w, t3b, ta, B, 128, 256, 32);
  k_conv1d<<<GRID(524288), 256, 0, stream>>>(ta, t4w, (const float*)nullptr, tb, B, 256, 256, 32);
  k_bnstats<float><<<256, 256, 0, stream>>>(tb, B, 256, 32, meanb, istdb);
  k_bnapply<float><<<GRID(524288), 256, 0, stream>>>(tb, g256, b256, meanb, istdb, 256, 32, 524288);

  // pack gate weights once
  k_prepack_gatesw<<<GRID(655360), 256, 0, stream>>>(wih, whh, wxg, whg, Wgp);

  // --- sequential attention scan, 32 steps ---
  for (int t = 0; t < 32; ++t) {
    k_pack_frag<<<GRID(32768), 256, 0, stream>>>(tb, hbuf, Af, t, B);
    k_gates_wmma<<<40, 256, 0, stream>>>(Af, Wgp, bih, bhh, bxg, bhg, G);
    k_state<<<GRID(16384), 256, 0, stream>>>(G, cbuf, hnew, sent, B);
    k_hp<<<GRID(32768), 256, 0, stream>>>(hnew, wh, bh, hp, B);
    k_sz<<<64, 256, 0, stream>>>(sent, wsw, bs, hp, wo, bo, szb, B);
    k_zatt<<<4096, 64, 0, stream>>>(vproj, hp, wo, bo, zb, B);
    k_attend<<<64, 256, 0, stream>>>(zb, szb, V, sent, hnew, hbuf, ctxs, B);
  }

  // --- heads ---
  k_shapeout<<<GRID(8192), 256, 0, stream>>>(ctxs, sow, sob, out_f, B);
  k_texthidden<<<GRID(16384), 256, 0, stream>>>(hbuf, tow1, tob1, th, B);
  k_textout<<<GRID(8192), 256, 0, stream>>>(th, tow2, tob2, out_f + 8192, B);
}
